// GCN_12403865551655
// MI455X (gfx1250) — matrix operand order repaired and checked by reading, not compiled, not run on hardware
//
#include <hip/hip_runtime.h>
#include <hip/hip_bf16.h>

typedef __attribute__((ext_vector_type(16))) _Float16 v16h;
typedef __attribute__((ext_vector_type(8)))  float    v8f;

#define N_NODES 50000
#define N_EDGES 800000
#define IN_FEATS 64
#define H_FEATS 128
#define NUM_CLASSES 32
#define TILES_M (N_NODES / 16)   // 3125, exact

// ---------------------------------------------------------------------------
// Utility kernels
// ---------------------------------------------------------------------------
__global__ __launch_bounds__(256) void zero_kernel(float* __restrict__ p, int n) {
    int t = blockIdx.x * blockDim.x + threadIdx.x;
    if (t < n) p[t] = 0.0f;
}

__global__ __launch_bounds__(256) void degree_kernel(const int* __restrict__ src,
                                                     const int* __restrict__ dst,
                                                     float* __restrict__ deg_out,
                                                     float* __restrict__ deg_in, int E) {
    int t = blockIdx.x * blockDim.x + threadIdx.x;
    if (t < E) {
        atomicAdd(&deg_out[src[t]], 1.0f);
        atomicAdd(&deg_in[dst[t]], 1.0f);
    }
}

__global__ __launch_bounds__(256) void rsqrt_kernel(float* __restrict__ a,
                                                    float* __restrict__ b, int n) {
    int t = blockIdx.x * blockDim.x + threadIdx.x;
    if (t < n) {
        a[t] = rsqrtf(fmaxf(a[t], 1.0f));
        b[t] = rsqrtf(fmaxf(b[t], 1.0f));
    }
}

// ---------------------------------------------------------------------------
// Pre-swizzle a weight matrix W[128, ncols] into f16 WMMA B-fragments.
// Fragment index: ((nt*4 + kch)*32 + lane); each fragment = 16 contiguous
// halves (32 B, v16h-load aligned). Element i of a fragment:
//   W[(kch*32 + (lane>=16 ? 8 : 0) + i + (i>=8 ? 8 : 0)) * ncols + nt*16 + (lane&15)]
// ---------------------------------------------------------------------------
__global__ __launch_bounds__(256) void packW_kernel(const float* __restrict__ W,
                                                    _Float16* __restrict__ P,
                                                    int ncols, int ntiles) {
    int t = blockIdx.x * blockDim.x + threadIdx.x;   // one thread per fragment
    if (t >= ntiles * 4 * 32) return;
    const int lane = t & 31;
    const int kch  = (t >> 5) & 3;
    const int nt   = t >> 7;
    const int nc = nt * 16 + (lane & 15);
    const int kb = kch * 32 + ((lane >= 16) ? 8 : 0);
    _Float16* dst = P + t * 16;
#pragma unroll
    for (int i = 0; i < 16; ++i)
        dst[i] = (_Float16)W[(kb + i + (i >= 8 ? 8 : 0)) * ncols + nc];
}

// ---------------------------------------------------------------------------
// GEMM1: X1[n, 128] = (concat(feat1, feat2)[n,:] * inv_out[n]) @ W1[128,128]
// One wave32 per 16x16 tile; block = 8 waves = one 16-row strip x 128 cols.
// B fragments come pre-swizzled in f16 (one v16h load per k-step).
// ---------------------------------------------------------------------------
__global__ __launch_bounds__(256) void gemm1_wmma_kernel(
    const float* __restrict__ feat1, const float* __restrict__ feat2,
    const float* __restrict__ inv_out, const _Float16* __restrict__ PW1,
    float* __restrict__ X1) {
    const int wave = threadIdx.x >> 5;
    const int lane = threadIdx.x & 31;
    const int tile_m = blockIdx.x;                 // 0..3124
    const int n0 = wave * 16;                      // column tile base
    const bool hi = lane >= 16;
    const int m_row = (tile_m << 4) + (lane & 15); // A row this lane feeds
    const int ncol = n0 + (lane & 15);             // C column this lane owns
    const float scale = inv_out[m_row];
    const _Float16* pw = PW1 + ((wave * 4) * 32 + lane) * 16;

    v8f c = {};
#pragma unroll
    for (int kch = 0; kch < 4; ++kch) {
        const int kc = kch * 32;
        // Source matrix is uniform per k-chunk: kc<64 -> feat1, else feat2
        const float* __restrict__ F = (kc < IN_FEATS) ? feat1 : feat2;
        const int abase = (kc & (IN_FEATS - 1)) + (hi ? 8 : 0);
        v16h a;
#pragma unroll
        for (int i = 0; i < 8; ++i) {
            a[i]     = (_Float16)(F[m_row * IN_FEATS + abase + i] * scale);
            a[8 + i] = (_Float16)(F[m_row * IN_FEATS + abase + 16 + i] * scale);
        }
        const v16h b = *(const v16h*)(pw + kch * 32 * 16);
        c = __builtin_amdgcn_wmma_f32_16x16x32_f16(false, a, false, b,
                                                   (short)0, c, false, false);
    }
    // C/D layout: VGPR r -> M = r (lanes 0-15) / 8+r (lanes 16-31)
    const int mbase = (tile_m << 4) + (hi ? 8 : 0);
#pragma unroll
    for (int r = 0; r < 8; ++r)
        X1[(mbase + r) * H_FEATS + ncol] = c[r];
}

// ---------------------------------------------------------------------------
// GEMM2: X2[n, 32] = (H1[n,:] * inv_out[n]) @ W2[128,32]
// 8 waves per block cover 4 row tiles x 2 col tiles.
// ---------------------------------------------------------------------------
__global__ __launch_bounds__(256) void gemm2_wmma_kernel(
    const float* __restrict__ H1, const float* __restrict__ inv_out,
    const _Float16* __restrict__ PW2, float* __restrict__ X2) {
    const int wave = threadIdx.x >> 5;
    const int lane = threadIdx.x & 31;
    const int tile_m = blockIdx.x * 4 + (wave >> 1);
    if (tile_m >= TILES_M) return;                 // whole-wave exit (EXEC stays full)
    const int nt = wave & 1;
    const bool hi = lane >= 16;
    const int m_row = (tile_m << 4) + (lane & 15);
    const int ncol = nt * 16 + (lane & 15);
    const float scale = inv_out[m_row];
    const _Float16* pw = PW2 + ((nt * 4) * 32 + lane) * 16;

    v8f c = {};
#pragma unroll
    for (int kch = 0; kch < 4; ++kch) {
        const int abase = kch * 32 + (hi ? 8 : 0);
        v16h a;
#pragma unroll
        for (int i = 0; i < 8; ++i) {
            a[i]     = (_Float16)(H1[m_row * H_FEATS + abase + i] * scale);
            a[8 + i] = (_Float16)(H1[m_row * H_FEATS + abase + 16 + i] * scale);
        }
        const v16h b = *(const v16h*)(pw + kch * 32 * 16);
        c = __builtin_amdgcn_wmma_f32_16x16x32_f16(false, a, false, b,
                                                   (short)0, c, false, false);
    }
    const int mbase = (tile_m << 4) + (hi ? 8 : 0);
#pragma unroll
    for (int r = 0; r < 8; ++r)
        X2[(mbase + r) * NUM_CLASSES + ncol] = c[r];
}

// ---------------------------------------------------------------------------
// Edge scatter-add: agg[dst[e], :] += X[src[e], :]  (float4 gather, f32 atomics)
// Fq = feats/4 per edge; one thread handles 4 features of one edge.
// ---------------------------------------------------------------------------
__global__ __launch_bounds__(256) void scatter_kernel(
    const float* __restrict__ X, const int* __restrict__ src,
    const int* __restrict__ dst, float* __restrict__ agg, int E, int Fq) {
    long long t = (long long)blockIdx.x * blockDim.x + threadIdx.x;
    long long total = (long long)E * Fq;
    if (t >= total) return;
    int e = (int)(t / Fq);
    int f = (int)(t % Fq) * 4;
    int s = src[e], d = dst[e];
    const float4 v = *(const float4*)(X + (long long)s * (Fq * 4) + f);
    float* p = agg + (long long)d * (Fq * 4) + f;
    atomicAdd(p + 0, v.x);
    atomicAdd(p + 1, v.y);
    atomicAdd(p + 2, v.z);
    atomicAdd(p + 3, v.w);
}

// H1[n,f] = relu(agg[n,f] * inv_in[n] + b1[f])
__global__ __launch_bounds__(256) void finish1_kernel(
    const float* __restrict__ agg, const float* __restrict__ inv_in,
    const float* __restrict__ b1, float* __restrict__ H1) {
    int t = blockIdx.x * blockDim.x + threadIdx.x;
    if (t >= N_NODES * H_FEATS) return;
    int n = t >> 7;        // /128
    int f = t & 127;
    float v = agg[t] * inv_in[n] + b1[f];
    H1[t] = fmaxf(v, 0.0f);
}

// acc[c] += sum over assigned rows of agg2[n,c] * inv_in[n]
__global__ __launch_bounds__(256) void reduce_kernel(
    const float* __restrict__ agg2, const float* __restrict__ inv_in,
    float* __restrict__ acc) {
    int c = threadIdx.x & 31;
    int row = blockIdx.x * 8 + (threadIdx.x >> 5);
    int stride = gridDim.x * 8;
    float s = 0.0f;
    for (int n = row; n < N_NODES; n += stride)
        s += agg2[n * NUM_CLASSES + c] * inv_in[n];
    atomicAdd(&acc[c], s);
}

__global__ void final_kernel(const float* __restrict__ acc,
                             const float* __restrict__ b2, float* __restrict__ out) {
    int c = threadIdx.x;
    if (c < NUM_CLASSES)
        out[c] = acc[c] * (1.0f / (float)N_NODES) + b2[c];
}

// ---------------------------------------------------------------------------
extern "C" void kernel_launch(void* const* d_in, const int* in_sizes, int n_in,
                              void* d_out, int out_size, void* d_ws, size_t ws_size,
                              hipStream_t stream) {
    const float* feat1 = (const float*)d_in[0];
    const float* feat2 = (const float*)d_in[1];
    const int*   src   = (const int*)d_in[2];
    const int*   dst   = (const int*)d_in[3];
    const float* W1    = (const float*)d_in[4];
    const float* b1    = (const float*)d_in[5];
    const float* W2    = (const float*)d_in[6];
    const float* b2    = (const float*)d_in[7];
    float* out = (float*)d_out;

    float* w       = (float*)d_ws;
    float* inv_out = w;                          // N
    float* inv_in  = w + N_NODES;                // N
    float* bufA    = w + 2 * N_NODES;            // N*128 : X1 -> H1 -> agg2
    float* bufB    = bufA + N_NODES * H_FEATS;   // N*128 : agg1 -> X2
    float* acc     = bufB + N_NODES * H_FEATS;   // 32
    _Float16* PW1  = (_Float16*)(acc + 32);      // 8*4*32*16 halves (32 KB)
    _Float16* PW2  = PW1 + 8 * 4 * 32 * 16;      // 2*4*32*16 halves (8 KB)

    // 0) pre-swizzle weights into WMMA B-fragment layout (f16)
    packW_kernel<<<4, 256, 0, stream>>>(W1, PW1, H_FEATS, 8);
    packW_kernel<<<1, 256, 0, stream>>>(W2, PW2, NUM_CLASSES, 2);

    // 1) degrees -> inv sqrt norms
    zero_kernel<<<(2 * N_NODES + 255) / 256, 256, 0, stream>>>(inv_out, 2 * N_NODES);
    degree_kernel<<<(N_EDGES + 255) / 256, 256, 0, stream>>>(src, dst, inv_out, inv_in, N_EDGES);
    rsqrt_kernel<<<(N_NODES + 255) / 256, 256, 0, stream>>>(inv_out, inv_in, N_NODES);

    // 2) layer 1 GEMM (WMMA): X1 = (concat * inv_out) @ W1
    gemm1_wmma_kernel<<<TILES_M, 256, 0, stream>>>(feat1, feat2, inv_out, PW1, bufA);

    // 3) edge scatter-add layer 1
    zero_kernel<<<(N_NODES * H_FEATS + 255) / 256, 256, 0, stream>>>(bufB, N_NODES * H_FEATS);
    {
        long long total = (long long)N_EDGES * (H_FEATS / 4);
        scatter_kernel<<<(int)((total + 255) / 256), 256, 0, stream>>>(
            bufA, src, dst, bufB, N_EDGES, H_FEATS / 4);
    }

    // 4) H1 = relu(agg1 * inv_in + b1)   (overwrites X1 buffer)
    finish1_kernel<<<(N_NODES * H_FEATS + 255) / 256, 256, 0, stream>>>(bufB, inv_in, b1, bufA);

    // 5) layer 2 GEMM (WMMA): X2 = (H1 * inv_out) @ W2   (into bufB)
    gemm2_wmma_kernel<<<(TILES_M + 3) / 4, 256, 0, stream>>>(bufA, inv_out, PW2, bufB);

    // 6) edge scatter-add layer 2 (agg2 into bufA)
    zero_kernel<<<(N_NODES * NUM_CLASSES + 255) / 256, 256, 0, stream>>>(bufA, N_NODES * NUM_CLASSES);
    {
        long long total = (long long)N_EDGES * (NUM_CLASSES / 4);
        scatter_kernel<<<(int)((total + 255) / 256), 256, 0, stream>>>(
            bufB, src, dst, bufA, N_EDGES, NUM_CLASSES / 4);
    }

    // 7) mean over nodes of (agg2 * inv_in) + b2
    zero_kernel<<<1, 256, 0, stream>>>(acc, NUM_CLASSES);
    reduce_kernel<<<256, 256, 0, stream>>>(bufA, inv_in, acc);
    final_kernel<<<1, 32, 0, stream>>>(acc, b2, out);
}
